// MultiHeadAttention_25675314495840
// MI455X (gfx1250) — matrix operand order repaired and checked by reading, not compiled, not run on hardware
//
#include <hip/hip_runtime.h>
#include <hip/hip_bf16.h>

// ---------------- problem constants ----------------
#define BATCH   4
#define SEQ     2048
#define DMODEL  1024
#define NHEADS  16
#define DKHEAD  64   // DMODEL / NHEADS

typedef __attribute__((ext_vector_type(16))) __bf16 v16bf;
typedef __attribute__((ext_vector_type(16))) float  v16f;
typedef __attribute__((ext_vector_type(8)))  float  v8f;

struct Bits256 { uint4 lo, hi; };
struct F32x16  { float4 a, b, c, d; };

__device__ __forceinline__ __bf16 f2bf(float f) {
  return static_cast<__bf16>(f);   // RNE; backend picks native cvt
}

__device__ __forceinline__ v16bf pack16(uint4 lo, uint4 hi) {
  Bits256 t{lo, hi};
  return __builtin_bit_cast(v16bf, t);
}

// ---- A fragment (16x32, 16-bit layout): lane m=lane&15, khalf=lane>>4.
// elements 0..7  -> K = k0 + khalf*8 + e        (16B contiguous)
// elements 8..15 -> K = k0 + 16 + khalf*8 + e   (16B contiguous)
__device__ __forceinline__ v16bf load_a_bf16(const __bf16* __restrict__ A,
                                             int m0, int k0, int lda, int lane) {
  int m = m0 + (lane & 15);
  int kh = (lane >> 4) & 1;
  const __bf16* p0 = A + (size_t)m * lda + k0 + kh * 8;
  __builtin_prefetch(p0 + 64, 0, 3);                   // near-scope: next K-tiles
  uint4 lo = *(const uint4*)p0;
  uint4 hi = *(const uint4*)(p0 + 16);
  return pack16(lo, hi);
}

__device__ __forceinline__ v16bf load_a_f32(const float* __restrict__ A,
                                            int m0, int k0, int lda, int lane) {
  int m = m0 + (lane & 15);
  int kh = (lane >> 4) & 1;
  const float* p0 = A + (size_t)m * lda + k0 + kh * 8;
  __builtin_prefetch(p0 + 64, 0, 3);
  F32x16 t;
  t.a = ((const float4*)p0)[0];
  t.b = ((const float4*)p0)[1];
  t.c = ((const float4*)(p0 + 16))[0];
  t.d = ((const float4*)(p0 + 16))[1];
  v16f f = __builtin_bit_cast(v16f, t);
  return __builtin_convertvector(f, v16bf);            // packed f32 -> bf16
}

// ---- B fragment from K-contiguous ("transposed") storage Bt[N][K]:
// lane n=lane&15, khalf=lane>>4; elements 0..7 -> K = k0 + khalf*8 + e,
// elements 8..15 -> K = k0 + 16 + khalf*8 + e (two 16B runs per lane).
__device__ __forceinline__ v16bf load_b_bf16(const __bf16* __restrict__ Bt,
                                             int n0, int k0, int ldb, int lane) {
  int n = n0 + (lane & 15);
  int kh = (lane >> 4) & 1;
  const __bf16* p = Bt + (size_t)n * ldb + k0 + kh * 8;
  __builtin_prefetch(p + 64, 0, 3);
  uint4 lo = *(const uint4*)p;
  uint4 hi = *(const uint4*)(p + 16);
  return pack16(lo, hi);
}

// ---------------- WMMA GEMM ----------------
// C[M,N] = A[M,K] * Bt[N,K]^T  (Bt stored K-contiguous per output column)
// Wave tile: 32(M) x 64(N) = 2x4 fragments, 8 f32 accumulators
// (6 fragment loads per 8 v_wmma -> 0.75 loads/WMMA).
// Block: BM x BN waves. Grid.z = batch*heads, split by H with per-b / per-h
// element strides for each operand.
template<int BM, int BN, bool AF32, bool CF32>
__global__ __launch_bounds__(BM * BN * 32)
void wmma_gemm(const void* __restrict__ Aptr, const __bf16* __restrict__ Btptr,
               void* __restrict__ Cptr, int K, int lda, int ldb, int ldc,
               size_t aB, size_t aH, size_t bB, size_t bH, size_t cB, size_t cH,
               int H, float scale) {
  const int z = blockIdx.z;
  const int b = z / H;
  const int h = z - b * H;

  const float*  Af  = (const float*)Aptr  + (AF32 ? ((size_t)b * aB + (size_t)h * aH) : 0);
  const __bf16* Abf = (const __bf16*)Aptr + (AF32 ? 0 : ((size_t)b * aB + (size_t)h * aH));
  const __bf16* Bt  = Btptr + (size_t)b * bB + (size_t)h * bH;
  float*  Cf  = (float*)Cptr   + (CF32 ? ((size_t)b * cB + (size_t)h * cH) : 0);
  __bf16* Cbf = (__bf16*)Cptr  + (CF32 ? 0 : ((size_t)b * cB + (size_t)h * cH));

  const int wave = threadIdx.x >> 5;
  const int lane = threadIdx.x & 31;
  const int wm = wave / BN;
  const int wn = wave - wm * BN;
  const int m0 = blockIdx.x * (32 * BM) + wm * 32;
  const int n0 = blockIdx.y * (64 * BN) + wn * 64;

  v8f acc[2][4] = {};

  for (int k0 = 0; k0 < K; k0 += 32) {
    v16bf a0, a1;
    if constexpr (AF32) {
      a0 = load_a_f32(Af, m0,      k0, lda, lane);
      a1 = load_a_f32(Af, m0 + 16, k0, lda, lane);
    } else {
      a0 = load_a_bf16(Abf, m0,      k0, lda, lane);
      a1 = load_a_bf16(Abf, m0 + 16, k0, lda, lane);
    }
    v16bf bf[4];
#pragma unroll
    for (int t = 0; t < 4; ++t)
      bf[t] = load_b_bf16(Bt, n0 + 16 * t, k0, ldb, lane);

#pragma unroll
    for (int t = 0; t < 4; ++t) {
      acc[0][t] = __builtin_amdgcn_wmma_f32_16x16x32_bf16(
          false, a0, false, bf[t], (short)0, acc[0][t], false, false);
      acc[1][t] = __builtin_amdgcn_wmma_f32_16x16x32_bf16(
          false, a1, false, bf[t], (short)0, acc[1][t], false, false);
    }
  }

  // C/D layout: lane n = lane&15; row = frag_m*16 + 8*(lane>>4) + vgpr
  const int nn = lane & 15;
  const int kh = lane >> 4;
#pragma unroll
  for (int i = 0; i < 2; ++i)
#pragma unroll
    for (int t = 0; t < 4; ++t)
#pragma unroll
      for (int e = 0; e < 8; ++e) {
        int r = m0 + 16 * i + 8 * kh + e;
        int c = n0 + 16 * t + nn;
        float v = acc[i][t][e] * scale;
        if constexpr (CF32) Cf[(size_t)r * ldc + c] = v;
        else                Cbf[(size_t)r * ldc + c] = f2bf(v);
      }
}

// ---------------- transpose + convert kernels ----------------
// out[c][r] = (bf16) in[r][c]   (fp32 weights -> K-contiguous bf16)
__global__ __launch_bounds__(256)
void transpose_f32_to_bf16(const float* __restrict__ in, __bf16* __restrict__ out,
                           int rows, int cols) {
  __shared__ float tile[32][33];
  int c0 = blockIdx.x * 32, r0 = blockIdx.y * 32;
  int tx = threadIdx.x, ty = threadIdx.y;   // 32 x 8
#pragma unroll
  for (int i = 0; i < 32; i += 8)
    tile[ty + i][tx] = in[(size_t)(r0 + ty + i) * cols + c0 + tx];
  __syncthreads();
#pragma unroll
  for (int i = 0; i < 32; i += 8)
    out[(size_t)(c0 + ty + i) * rows + r0 + tx] = f2bf(tile[tx][ty + i]);
}

// per-batch bf16 transpose: out[b][c][r] = in[b][r][c]
__global__ __launch_bounds__(256)
void transpose_bf16(const __bf16* __restrict__ in, __bf16* __restrict__ out,
                    int rows, int cols) {
  const __bf16* src = in  + (size_t)blockIdx.z * rows * cols;
  __bf16*       dst = out + (size_t)blockIdx.z * rows * cols;
  __shared__ __bf16 tile[32][33];
  int c0 = blockIdx.x * 32, r0 = blockIdx.y * 32;
  int tx = threadIdx.x, ty = threadIdx.y;
#pragma unroll
  for (int i = 0; i < 32; i += 8)
    tile[ty + i][tx] = src[(size_t)(r0 + ty + i) * cols + c0 + tx];
  __syncthreads();
#pragma unroll
  for (int i = 0; i < 32; i += 8)
    dst[(size_t)(c0 + ty + i) * rows + r0 + tx] = tile[tx][ty + i];
}

// ---------------- row softmax over 2048 elements, in place ----------------
__global__ __launch_bounds__(256)
void softmax2048(float* __restrict__ attn) {
  float* row = attn + (size_t)blockIdx.x * 2048;
  const int t = threadIdx.x;
  const int w = t >> 5, l = t & 31;
  float4* rv = (float4*)row;
  float4 a = rv[2 * t];
  float4 c = rv[2 * t + 1];

  __shared__ float red[8];

  float mx = fmaxf(fmaxf(fmaxf(a.x, a.y), fmaxf(a.z, a.w)),
                   fmaxf(fmaxf(c.x, c.y), fmaxf(c.z, c.w)));
#pragma unroll
  for (int o = 16; o > 0; o >>= 1) mx = fmaxf(mx, __shfl_xor(mx, o, 32));
  if (l == 0) red[w] = mx;
  __syncthreads();
  mx = red[0];
#pragma unroll
  for (int i = 1; i < 8; ++i) mx = fmaxf(mx, red[i]);
  __syncthreads();

  a.x = __expf(a.x - mx); a.y = __expf(a.y - mx);
  a.z = __expf(a.z - mx); a.w = __expf(a.w - mx);
  c.x = __expf(c.x - mx); c.y = __expf(c.y - mx);
  c.z = __expf(c.z - mx); c.w = __expf(c.w - mx);

  float s = (a.x + a.y) + (a.z + a.w) + (c.x + c.y) + (c.z + c.w);
#pragma unroll
  for (int o = 16; o > 0; o >>= 1) s += __shfl_xor(s, o, 32);
  if (l == 0) red[w] = s;
  __syncthreads();
  float tot = red[0];
#pragma unroll
  for (int i = 1; i < 8; ++i) tot += red[i];
  float inv = 1.0f / tot;

  a.x *= inv; a.y *= inv; a.z *= inv; a.w *= inv;
  c.x *= inv; c.y *= inv; c.z *= inv; c.w *= inv;
  rv[2 * t] = a;
  rv[2 * t + 1] = c;
}

// ---------------- host-side pipeline ----------------
extern "C" void kernel_launch(void* const* d_in, const int* in_sizes, int n_in,
                              void* d_out, int out_size, void* d_ws, size_t ws_size,
                              hipStream_t stream) {
  const float* q   = (const float*)d_in[0];
  const float* k   = (const float*)d_in[1];
  const float* v   = (const float*)d_in[2];
  const float* w_q = (const float*)d_in[3];
  const float* w_k = (const float*)d_in[4];
  const float* w_v = (const float*)d_in[5];
  const float* w_o = (const float*)d_in[6];

  float* out  = (float*)d_out;                                   // [B,S,D]
  float* attn = out + (size_t)BATCH * SEQ * DMODEL;              // [B,H,S,S]

  // workspace layout (bf16 buffers), ~88 MB total
  char* ws = (char*)d_ws;
  size_t off = 0;
  const size_t W_ELEMS = (size_t)DMODEL * DMODEL;      // 1M
  const size_t P_ELEMS = (size_t)BATCH * SEQ * DMODEL; // 8M
  __bf16* wqt = (__bf16*)(ws + off); off += W_ELEMS * 2;
  __bf16* wkt = (__bf16*)(ws + off); off += W_ELEMS * 2;
  __bf16* wvt = (__bf16*)(ws + off); off += W_ELEMS * 2;
  __bf16* wot = (__bf16*)(ws + off); off += W_ELEMS * 2;
  __bf16* Qp  = (__bf16*)(ws + off); off += P_ELEMS * 2;
  __bf16* Kp  = (__bf16*)(ws + off); off += P_ELEMS * 2;
  __bf16* Vp  = (__bf16*)(ws + off); off += P_ELEMS * 2;
  __bf16* Vpt = (__bf16*)(ws + off); off += P_ELEMS * 2;   // [B][D][S]
  __bf16* ctx = (__bf16*)(ws + off); off += P_ELEMS * 2;

  const int M = BATCH * SEQ;          // 8192
  dim3 blk(256);
  dim3 tblk(32, 8);

  // 1) weights: fp32 [K,N] -> bf16 transposed [N,K]
  dim3 wt_grid(DMODEL / 32, DMODEL / 32);
  transpose_f32_to_bf16<<<wt_grid, tblk, 0, stream>>>(w_q, wqt, DMODEL, DMODEL);
  transpose_f32_to_bf16<<<wt_grid, tblk, 0, stream>>>(w_k, wkt, DMODEL, DMODEL);
  transpose_f32_to_bf16<<<wt_grid, tblk, 0, stream>>>(w_v, wvt, DMODEL, DMODEL);
  transpose_f32_to_bf16<<<wt_grid, tblk, 0, stream>>>(w_o, wot, DMODEL, DMODEL);

  // 2) projections: fp32 input @ bf16 weight^T -> bf16 [M, D]
  dim3 pg(M / 64, DMODEL / 256, 1);
  wmma_gemm<2, 4, true, false><<<pg, blk, 0, stream>>>(
      q, wqt, Qp, DMODEL, DMODEL, DMODEL, DMODEL,
      0, 0, 0, 0, 0, 0, 1, 1.0f);
  wmma_gemm<2, 4, true, false><<<pg, blk, 0, stream>>>(
      k, wkt, Kp, DMODEL, DMODEL, DMODEL, DMODEL,
      0, 0, 0, 0, 0, 0, 1, 1.0f);
  wmma_gemm<2, 4, true, false><<<pg, blk, 0, stream>>>(
      v, wvt, Vp, DMODEL, DMODEL, DMODEL, DMODEL,
      0, 0, 0, 0, 0, 0, 1, 1.0f);

  // 3) Vp [b][S][D] -> Vpt [b][D][S]  (K-contiguous B operand for PV GEMM)
  dim3 vt_grid(DMODEL / 32, SEQ / 32, BATCH);
  transpose_bf16<<<vt_grid, tblk, 0, stream>>>(Vp, Vpt, SEQ, DMODEL);

  // 4) scores = Qh @ Kh^T * 1/sqrt(dk) -> attn (fp32).  Kp rows are already
  //    K-contiguous for the transposed product.
  {
    dim3 g(SEQ / 64, SEQ / 256, BATCH * NHEADS);
    wmma_gemm<2, 4, false, true><<<g, blk, 0, stream>>>(
        Qp, Kp, attn, DKHEAD, DMODEL, DMODEL, SEQ,
        (size_t)SEQ * DMODEL, (size_t)DKHEAD,              // A: per-b, per-h
        (size_t)SEQ * DMODEL, (size_t)DKHEAD,              // Bt
        (size_t)NHEADS * SEQ * SEQ, (size_t)SEQ * SEQ,     // C
        NHEADS, 0.125f);                                   // 1/sqrt(64)
  }

  // 5) softmax over rows of attn, in place
  softmax2048<<<dim3(BATCH * NHEADS * SEQ), blk, 0, stream>>>(attn);

  // 6) ctx_h = attn @ Vh : A = fp32 attn (converted in-fragment), Bt = Vpt
  {
    dim3 g(SEQ / 256, DKHEAD / 64, BATCH * NHEADS);
    wmma_gemm<8, 1, true, false><<<g, blk, 0, stream>>>(
        attn, Vpt, ctx, SEQ, SEQ, SEQ, DMODEL,
        (size_t)NHEADS * SEQ * SEQ, (size_t)SEQ * SEQ,     // A
        (size_t)DMODEL * SEQ, (size_t)DKHEAD * SEQ,        // Bt (Vpt head rows)
        (size_t)SEQ * DMODEL, (size_t)DKHEAD,              // C (head columns)
        NHEADS, 1.0f);
  }

  // 7) out = ctx @ w_o -> fp32
  wmma_gemm<2, 4, false, true><<<pg, blk, 0, stream>>>(
      ctx, wot, out, DMODEL, DMODEL, DMODEL, DMODEL,
      0, 0, 0, 0, 0, 0, 1, 1.0f);

  (void)in_sizes; (void)n_in; (void)out_size; (void)ws_size;
}
